// TransformerEncoderReadout_solv_10222022164949
// MI455X (gfx1250) — compile-verified, matrix operand order checked by reading, not run on hardware
//
#include <hip/hip_runtime.h>
#include <hip/hip_bf16.h>

typedef __attribute__((ext_vector_type(16))) _Float16 v16h;
typedef __attribute__((ext_vector_type(8)))  _Float16 h8;
typedef __attribute__((ext_vector_type(8)))  float    v8f;
typedef __attribute__((ext_vector_type(4)))  unsigned int u32x4;
typedef __attribute__((ext_vector_type(8)))  int          i32x8;
typedef __attribute__((ext_vector_type(4)))  int          i32x4;

#define B_  256
#define S_  160
#define D_  128
#define H_  8
#define F_  256
#define NW  10
#define TPB (NW * 32)

// ---------------- WMMA helpers (CDNA5 16x16x32 f16 -> f32) ----------------

__device__ __forceinline__ v8f wmma32(v16h a, v16h b, v8f c) {
  return __builtin_amdgcn_wmma_f32_16x16x32_f16(false, a, false, b, (short)0, c,
                                                false, false);
}

// A fragment: 16x32 tile of row-major f16 matrix (K contiguous).
// Per ISA 7.12.2 the lane's 16 halves are two contiguous 8-half runs:
//   k = kb..kb+7 and 16+kb..16+kb+7, kb = (lane&16)?8:0, row m = lane&15.
__device__ __forceinline__ v16h load_a_frag(const _Float16* A, int lda, int lane) {
  const int m  = lane & 15;
  const int kb = (lane & 16) ? 8 : 0;
  const _Float16* p = A + m * lda + kb;
  h8 lo = *(const h8*)p;          // ds_load_b128 / global_load_b128
  h8 hi = *(const h8*)(p + 16);
  v16h r;
#pragma unroll
  for (int i = 0; i < 8; ++i) { r[i] = lo[i]; r[i + 8] = hi[i]; }
  return r;
}

// B fragment from a K-major (transposed) source: src[n][k], leading dim lds.
// B[k][n] = src[n*lds + k]; same two contiguous 8-half runs per lane (n = lane&15).
__device__ __forceinline__ v16h load_bT_frag(const _Float16* Sm, int lds, int lane) {
  const int n  = lane & 15;
  const int kb = (lane & 16) ? 8 : 0;
  const _Float16* p = Sm + n * lds + kb;
  h8 lo = *(const h8*)p;
  h8 hi = *(const h8*)(p + 16);
  v16h r;
#pragma unroll
  for (int i = 0; i < 8; ++i) { r[i] = lo[i]; r[i + 8] = hi[i]; }
  return r;
}

// C/D fragment store as f16, row-major (scattered b16 per lane-column)
__device__ __forceinline__ void store_c_f16(_Float16* C, int ldc, int lane, v8f c) {
  const int n  = lane & 15;
  const int mh = (lane & 16) ? 8 : 0;
#pragma unroll
  for (int j = 0; j < 8; ++j) C[(j + mh) * ldc + n] = (_Float16)c[j];
}

// C/D fragment store as f16 TRANSPOSED: CT[col][row]; per-lane rows are
// contiguous -> one 16-byte ds_store_b128.
__device__ __forceinline__ void store_c_f16_T(_Float16* CT, int ldct, int lane, v8f c) {
  const int n  = lane & 15;
  const int mh = (lane & 16) ? 8 : 0;
  h8 v;
#pragma unroll
  for (int j = 0; j < 8; ++j) v[j] = (_Float16)c[j];
  *(h8*)(CT + n * ldct + mh) = v;
}

__device__ __forceinline__ float redmax16(float v) {
#pragma unroll
  for (int m = 1; m <= 8; m <<= 1) v = fmaxf(v, __shfl_xor(v, m, 32));
  return v;
}
__device__ __forceinline__ float redsum16(float v) {
#pragma unroll
  for (int m = 1; m <= 8; m <<= 1) v += __shfl_xor(v, m, 32);
  return v;
}

// ---------------- Tensor Data Mover: 2D f16 tile -> LDS ----------------
// D# per CDNA5 ISA. dims in data_size units (2B).
// This toolchain exposes the 6-arg builtin:
//   (u32x4 g0, i32x8 g1, i32x4 g2, i32x4 g3, i32x8 extra, i32 cpol)

__device__ __forceinline__ void tdm_load_2d_f16(unsigned lds_byte_off, const void* gaddr,
                                                unsigned dim0, unsigned dim1,
                                                unsigned stride0) {
#if __has_builtin(__builtin_amdgcn_tensor_load_to_lds)
  const unsigned long long ga = (unsigned long long)gaddr;
  u32x4 g0;
  g0[0] = 1u;                                   // count=1 (valid user descriptor)
  g0[1] = lds_byte_off;                         // lds_addr
  g0[2] = (unsigned)(ga & 0xffffffffu);         // global_addr[31:0]
  g0[3] = (unsigned)((ga >> 32) & 0x01ffffffu)  // global_addr[56:32]
        | (2u << 30);                           // type = 2 ("image")
  i32x8 g1;
  g1[0] = (int)(1u << 16);                      // workgroup_mask=0, data_size=1 (2B)
  g1[1] = (int)(dim0 << 16);                    // tensor_dim0[15:0] @ bits63:48
  g1[2] = (int)((dim0 >> 16) | (dim1 << 16));   // tensor_dim0[31:16] | tensor_dim1[15:0]
  g1[3] = (int)((dim1 >> 16) | (dim0 << 16));   // tensor_dim1[31:16] | tile_dim0 (=dim0)
  g1[4] = (int)(dim1 & 0xffffu);                // tile_dim1 (=dim1), tile_dim2=0
  g1[5] = (int)stride0;                         // tensor_dim0_stride[31:0]
  g1[6] = 0;                                    // stride0[47:32], dim1_stride[15:0] (2D: unused)
  g1[7] = 0;
  i32x4 z4 = {};
  i32x8 z8 = {};
  __builtin_amdgcn_tensor_load_to_lds(g0, g1, z4, z4, z8, 0);
#endif
}

__device__ __forceinline__ void tdm_wait0() {
#if __has_builtin(__builtin_amdgcn_s_wait_tensorcnt)
  __builtin_amdgcn_s_wait_tensorcnt(0);
#endif
}

// ---------------- prep kernels ----------------

__global__ void k_zero(float* x, _Float16* xh, int* counts, int total) {
  int i = blockIdx.x * blockDim.x + threadIdx.x;
  if (i < total) { x[i] = 0.0f; xh[i] = (_Float16)0.0f; }
  if (i < B_) counts[i] = 0;
}

__global__ void k_count(const int* __restrict__ mol, int N, int* counts) {
  int i = blockIdx.x * blockDim.x + threadIdx.x;
  if (i < N) atomicAdd(&counts[mol[i]], 1);
}

__global__ void k_scan(const int* __restrict__ counts, int* starts) {
  if (threadIdx.x == 0) {
    int acc = 0;
    for (int i = 0; i < B_; ++i) { starts[i] = acc; acc += counts[i]; }
  }
}

__global__ void k_scatter(const float* __restrict__ atoms, const int* __restrict__ mol,
                          const int* __restrict__ starts, float* x, _Float16* xh, int N) {
  int row = blockIdx.x;
  if (row >= N) return;
  int b   = mol[row];
  int pos = row - starts[b];
  float v = atoms[(size_t)row * D_ + threadIdx.x];
  size_t o = ((size_t)b * S_ + pos) * D_ + threadIdx.x;
  x[o]  = v;
  xh[o] = (_Float16)v;
}

// All B-operands are stored K-major (transposed) for b128 fragment loads.
__global__ void k_weights(const float* __restrict__ wq, const float* __restrict__ wk,
                          const float* __restrict__ wv, const float* __restrict__ wo,
                          const float* __restrict__ w1, const float* __restrict__ w2,
                          _Float16* wqkvT,   // [3072][128]  (col-major of [D,3072])
                          _Float16* wohT,    // [128][1024]  (col-major of [1024,D])
                          _Float16* w1T,     // [256][128]
                          _Float16* w2T) {   // [128][256]
  const int n_qkv = D_ * 3072;
  const int n_wo  = 1024 * D_;
  const int n_w1  = D_ * F_;
  const int n_w2  = F_ * D_;
  int i = blockIdx.x * blockDim.x + threadIdx.x;
  if (i < n_qkv) {
    int r = i / 3072, c = i % 3072;
    float v = (c < 1024) ? wq[r * 1024 + c]
            : (c < 2048) ? wk[r * 1024 + (c - 1024)]
                         : wv[r * 1024 + (c - 2048)];
    wqkvT[c * D_ + r] = (_Float16)v;
  } else if (i < n_qkv + n_wo) {
    int t = i - n_qkv;
    int hk = t / D_, d = t % D_;
    wohT[d * 1024 + hk] = (_Float16)wo[t];
  } else if (i < n_qkv + n_wo + n_w1) {
    int t = i - n_qkv - n_wo;
    int d = t / F_, f = t % F_;
    w1T[f * D_ + d] = (_Float16)w1[t];
  } else if (i < n_qkv + n_wo + n_w1 + n_w2) {
    int t = i - n_qkv - n_wo - n_w1;
    int f = t / D_, d = t % D_;
    w2T[d * F_ + f] = (_Float16)w2[t];
  }
}

// ---------------- fused encoder: one workgroup per molecule ----------------

__global__ __launch_bounds__(TPB) void encoder_kernel(
    const float* __restrict__ x,        // [B,S,D] f32 dense padded
    const _Float16* __restrict__ xh,    // [B,S,D] f16
    const _Float16* __restrict__ wqkvT, // [3072][128]
    const float* __restrict__ bq, const float* __restrict__ bk, const float* __restrict__ bv,
    const _Float16* __restrict__ wohT,  // [128][1024]
    const float* __restrict__ bo,
    const _Float16* __restrict__ w1T,   // [256][128]
    const float* __restrict__ b1,
    const _Float16* __restrict__ w2T,   // [128][256]
    const float* __restrict__ b2,
    const float* __restrict__ g1, const float* __restrict__ be1,
    const float* __restrict__ g2, const float* __restrict__ be2,
    const int* __restrict__ counts,
    float* __restrict__ out)            // [B, D]
{
  extern __shared__ char smem[];
  _Float16* sX  = (_Float16*)smem;        // 160x128 row-major
  _Float16* sQ  = sX + S_ * D_;           // 160x128 row-major
  _Float16* sK  = sQ + S_ * D_;           // 160x128 row-major (read K-major by Q.K^T)
  _Float16* sVT = sK + S_ * D_;           // 128x160 V transposed (K-major for P.V)
  _Float16* sP  = sVT + D_ * S_;          // 160x160 softmax probs
  _Float16* sO  = sP + S_ * S_;           // 160x128 per-head attn output
  float* sRed   = (float*)(sO + S_ * D_); // 128 readout accumulator

  const int b    = blockIdx.x;
  const int tid  = threadIdx.x;
  const int lane = tid & 31;
  const int wv   = tid >> 5;              // wave 0..9 owns rows [16w, 16w+16)
  const int row0 = wv * 16;
  const int cnt  = counts[b];
  const int n    = lane & 15;
  const int mh   = (lane & 16) ? 8 : 0;

  if (tid < D_) sRed[tid] = 0.0f;

  // --- stage x[b] into LDS via the Tensor Data Mover (wave 0 issues) ---
#if __has_builtin(__builtin_amdgcn_tensor_load_to_lds)
  if (wv == 0) {
    tdm_load_2d_f16(__builtin_amdgcn_groupstaticsize() /* sX @ dyn-LDS base */,
                    xh + (size_t)b * S_ * D_, D_, S_, D_);
    tdm_wait0();
  }
#else
  {
    const uint4* src = (const uint4*)(xh + (size_t)b * S_ * D_);
    uint4* dst = (uint4*)sX;
    for (int i = tid; i < S_ * D_ / 8; i += TPB) dst[i] = src[i];
  }
#endif
  __syncthreads();

  v8f accA[8];
#pragma unroll
  for (int c = 0; c < 8; ++c) accA[c] = {};

  const float scl = 0.08838834764831845f; // 1/sqrt(128)

  for (int h = 0; h < H_; ++h) {
    // --- Q,K,V projection for this wave's row tile ---
    for (int mtx = 0; mtx < 3; ++mtx) {
      _Float16* dst     = (mtx == 0) ? sQ : sK;
      const float* bias = (mtx == 0) ? bq : (mtx == 1) ? bk : bv;
      for (int c = 0; c < 8; ++c) {
        v8f acc = {};
#pragma unroll
        for (int kc = 0; kc < 4; ++kc) {
          v16h a  = load_a_frag(sX + row0 * D_ + kc * 32, D_, lane);
          v16h bb = load_bT_frag(wqkvT + (size_t)(mtx * 1024 + h * 128 + c * 16) * D_ + kc * 32,
                                 D_, lane);
          acc = wmma32(a, bb, acc);
        }
        float bvl = bias[h * 128 + c * 16 + n];
#pragma unroll
        for (int j = 0; j < 8; ++j) acc[j] += bvl;
        if (mtx == 2) store_c_f16_T(sVT + (c * 16) * S_ + row0, S_, lane, acc);
        else          store_c_f16(dst + row0 * D_ + c * 16, D_, lane, acc);
      }
    }
    __syncthreads();

    // --- scores = Q K^T / sqrt(K), key mask, softmax ---
    v8f st[10];
#pragma unroll
    for (int t = 0; t < 10; ++t) {
      v8f acc = {};
#pragma unroll
      for (int kc = 0; kc < 4; ++kc) {
        v16h a  = load_a_frag(sQ + row0 * D_ + kc * 32, D_, lane);
        v16h bb = load_bT_frag(sK + (t * 16) * D_ + kc * 32, D_, lane);
        acc = wmma32(a, bb, acc);
      }
      const float msk = ((t * 16 + n) < cnt) ? 0.0f : -1e9f;
#pragma unroll
      for (int j = 0; j < 8; ++j) acc[j] = acc[j] * scl + msk;
      st[t] = acc;
    }
    float rs[8];
#pragma unroll
    for (int j = 0; j < 8; ++j) {
      float m = -1e30f;
#pragma unroll
      for (int t = 0; t < 10; ++t) m = fmaxf(m, st[t][j]);
      m = redmax16(m);
      float s = 0.0f;
#pragma unroll
      for (int t = 0; t < 10; ++t) {
        float p = __expf(st[t][j] - m);
        st[t][j] = p;
        s += p;
      }
      rs[j] = redsum16(s);
    }
#pragma unroll
    for (int t = 0; t < 10; ++t)
#pragma unroll
      for (int j = 0; j < 8; ++j)
        sP[(row0 + j + mh) * S_ + t * 16 + n] = (_Float16)st[t][j];
    __syncthreads();

    // --- O = softmax(P) V (divide by row sum) ---
    for (int c = 0; c < 8; ++c) {
      v8f acc = {};
#pragma unroll
      for (int kc = 0; kc < 5; ++kc) {
        v16h a  = load_a_frag(sP + row0 * S_ + kc * 32, S_, lane);
        v16h bb = load_bT_frag(sVT + (c * 16) * S_ + kc * 32, S_, lane);
        acc = wmma32(a, bb, acc);
      }
#pragma unroll
      for (int j = 0; j < 8; ++j) acc[j] /= rs[j];
      store_c_f16(sO + row0 * D_ + c * 16, D_, lane, acc);
    }
    __syncthreads();

    // --- attn_out += O @ Wo[h] (accumulated across heads in registers) ---
#pragma unroll
    for (int c = 0; c < 8; ++c) {
#pragma unroll
      for (int kc = 0; kc < 4; ++kc) {
        v16h a  = load_a_frag(sO + row0 * D_ + kc * 32, D_, lane);
        v16h bb = load_bT_frag(wohT + (size_t)(c * 16) * 1024 + h * 128 + kc * 32, 1024, lane);
        accA[c] = wmma32(a, bb, accA[c]);
      }
    }
    __syncthreads();
  } // heads

  // --- + bo + residual x, LayerNorm1 ---
#pragma unroll
  for (int c = 0; c < 8; ++c) {
    const float bof = bo[c * 16 + n];
#pragma unroll
    for (int j = 0; j < 8; ++j) {
      const int row = row0 + j + mh;
      accA[c][j] += bof + x[((size_t)b * S_ + row) * D_ + c * 16 + n];
    }
  }
#pragma unroll
  for (int j = 0; j < 8; ++j) {
    float s = 0.0f;
#pragma unroll
    for (int c = 0; c < 8; ++c) s += accA[c][j];
    const float mean = redsum16(s) * (1.0f / 128.0f);
    float vs = 0.0f;
#pragma unroll
    for (int c = 0; c < 8; ++c) { float d = accA[c][j] - mean; vs += d * d; }
    const float rstd = rsqrtf(redsum16(vs) * (1.0f / 128.0f) + 1e-3f);
#pragma unroll
    for (int c = 0; c < 8; ++c) {
      const int col = c * 16 + n;
      accA[c][j] = (accA[c][j] - mean) * rstd * g1[col] + be1[col];
    }
  }
#pragma unroll
  for (int c = 0; c < 8; ++c) store_c_f16(sQ + row0 * D_ + c * 16, D_, lane, accA[c]);
  __syncthreads();

  // --- FFN1: relu(h1 @ w1 + b1) -> sF (160x256, reuses sK..sVT) ---
  _Float16* sF = sK;
  for (int c2 = 0; c2 < 16; ++c2) {
    v8f acc = {};
#pragma unroll
    for (int kc = 0; kc < 4; ++kc) {
      v16h a  = load_a_frag(sQ + row0 * D_ + kc * 32, D_, lane);
      v16h bb = load_bT_frag(w1T + (size_t)(c2 * 16) * D_ + kc * 32, D_, lane);
      acc = wmma32(a, bb, acc);
    }
    const float b1v = b1[c2 * 16 + n];
#pragma unroll
    for (int j = 0; j < 8; ++j) acc[j] = fmaxf(acc[j] + b1v, 0.0f);
    store_c_f16(sF + row0 * F_ + c2 * 16, F_, lane, acc);
  }
  __syncthreads();

  // --- FFN2 + residual + LayerNorm2 ---
  v8f o2[8];
#pragma unroll
  for (int c = 0; c < 8; ++c) {
    v8f acc = {};
#pragma unroll
    for (int kc = 0; kc < 8; ++kc) {
      v16h a  = load_a_frag(sF + row0 * F_ + kc * 32, F_, lane);
      v16h bb = load_bT_frag(w2T + (size_t)(c * 16) * F_ + kc * 32, F_, lane);
      acc = wmma32(a, bb, acc);
    }
    const float b2v = b2[c * 16 + n];
#pragma unroll
    for (int j = 0; j < 8; ++j) acc[j] += b2v + accA[c][j];
    o2[c] = acc;
  }
#pragma unroll
  for (int j = 0; j < 8; ++j) {
    float s = 0.0f;
#pragma unroll
    for (int c = 0; c < 8; ++c) s += o2[c][j];
    const float mean = redsum16(s) * (1.0f / 128.0f);
    float vs = 0.0f;
#pragma unroll
    for (int c = 0; c < 8; ++c) { float d = o2[c][j] - mean; vs += d * d; }
    const float rstd = rsqrtf(redsum16(vs) * (1.0f / 128.0f) + 1e-3f);
#pragma unroll
    for (int c = 0; c < 8; ++c) {
      const int col = c * 16 + n;
      o2[c][j] = (o2[c][j] - mean) * rstd * g2[col] + be2[col];
    }
  }

  // --- readout: column sums over all S rows, divide by true atom count ---
#pragma unroll
  for (int c = 0; c < 8; ++c) {
    float s = 0.0f;
#pragma unroll
    for (int j = 0; j < 8; ++j) s += o2[c][j];
    s += __shfl_xor(s, 16, 32);
    if (lane < 16) atomicAdd(&sRed[c * 16 + n], s);
  }
  __syncthreads();
  if (tid < D_) out[(size_t)b * D_ + tid] = sRed[tid] / (float)cnt;
}

// ---------------- host launch ----------------

extern "C" void kernel_launch(void* const* d_in, const int* in_sizes, int n_in,
                              void* d_out, int out_size, void* d_ws, size_t ws_size,
                              hipStream_t stream) {
  const float* atom = (const float*)d_in[0];
  const int*   mol  = (const int*)d_in[1];
  const float* wq = (const float*)d_in[2];
  const float* bq = (const float*)d_in[3];
  const float* wk = (const float*)d_in[4];
  const float* bk = (const float*)d_in[5];
  const float* wv = (const float*)d_in[6];
  const float* bv = (const float*)d_in[7];
  const float* wo = (const float*)d_in[8];
  const float* bo = (const float*)d_in[9];
  const float* w1 = (const float*)d_in[10];
  const float* b1 = (const float*)d_in[11];
  const float* w2 = (const float*)d_in[12];
  const float* b2 = (const float*)d_in[13];
  const float* g1 = (const float*)d_in[14];
  const float* be1 = (const float*)d_in[15];
  const float* g2 = (const float*)d_in[16];
  const float* be2 = (const float*)d_in[17];

  const int N = in_sizes[0] / D_;

  char* ws = (char*)d_ws;
  int* counts = (int*)ws;            ws += 1024;
  int* starts = (int*)ws;            ws += 1024;
  float*    x  = (float*)ws;         ws += (size_t)B_ * S_ * D_ * 4;
  _Float16* xh = (_Float16*)ws;      ws += (size_t)B_ * S_ * D_ * 2;
  _Float16* wqkvT = (_Float16*)ws;   ws += (size_t)D_ * 3072 * 2;
  _Float16* wohT  = (_Float16*)ws;   ws += (size_t)1024 * D_ * 2;
  _Float16* w1T   = (_Float16*)ws;   ws += (size_t)D_ * F_ * 2;
  _Float16* w2T   = (_Float16*)ws;   ws += (size_t)F_ * D_ * 2;

  const int total = B_ * S_ * D_;
  k_zero<<<(total + 255) / 256, 256, 0, stream>>>(x, xh, counts, total);
  k_count<<<(N + 255) / 256, 256, 0, stream>>>(mol, N, counts);
  k_scan<<<1, 32, 0, stream>>>(counts, starts);
  k_scatter<<<N, D_, 0, stream>>>(atom, mol, starts, x, xh, N);
  const int wtotal = D_ * 3072 + 1024 * D_ + D_ * F_ + F_ * D_;
  k_weights<<<(wtotal + 255) / 256, 256, 0, stream>>>(wq, wk, wv, wo, w1, w2,
                                                      wqkvT, wohT, w1T, w2T);

  // LDS: sX,sQ,sK,sVT,sO (5 * 160*128 f16) + sP (160*160 f16) + sRed (128 f32)
  const size_t lds = (size_t)(5 * S_ * D_ + S_ * S_) * 2 + 128 * 4; // 256512 B
  (void)hipFuncSetAttribute((const void*)encoder_kernel,
                            hipFuncAttributeMaxDynamicSharedMemorySize, (int)lds);
  encoder_kernel<<<B_, TPB, lds, stream>>>(x, xh, wqkvT, bq, bk, bv, wohT, bo,
                                           w1T, b1, w2T, b2, g1, be1, g2, be2,
                                           counts, (float*)d_out);
}
